// NSS_SSM_22393959481695
// MI455X (gfx1250) — compile-verified, matrix operand order checked by reading, not run on hardware
//
#include <hip/hip_runtime.h>
#include <hip/hip_bf16.h>

// ---------------------------------------------------------------------------
// VMamba-style SS2D forward for gfx1250 (MI455X).
// GEMMs: bf16 WMMA (v_wmma_f32_16x16x32_bf16), f32 accumulate.
// Selective scan: exact 3-phase chunked scan (64 chunks x 64 steps).
// ---------------------------------------------------------------------------

typedef __attribute__((ext_vector_type(16))) __bf16 v16bf;
typedef __attribute__((ext_vector_type(8)))  float  v8f;

#define DI    192      // d_inner
#define DM    96       // d_model
#define NS    16       // d_state
#define RK    6        // dt_rank
#define KD    4        // scan directions
#define BB    4        // batch
#define LL    4096     // H*W
#define CH    64       // number of chunks
#define CT    64       // chunk length
#define ROWS  16384    // B*L
#define XDLD  160      // padded x_dbl leading dim (38 -> 160/direction-packed)

__device__ __forceinline__ __bf16 f2bf(float f) {
  unsigned u = __float_as_uint(f);
  unsigned r = (u + 0x7FFFu + ((u >> 16) & 1u)) >> 16;
  unsigned short s = (unsigned short)r;
  return *reinterpret_cast<__bf16*>(&s);
}

union Frag { v16bf v; uint4 q[2]; };

// Load a 16-lane-group WMMA fragment from a row-major [rows x ld] bf16 matrix.
// Lane holds row (lane&15); lane group g = lane>>4 selects the K sub-stripes:
//   elements 0..7  -> k = k0 + 8g .. k0 + 8g + 7
//   elements 8..15 -> k = k0 + 16 + 8g .. k0 + 16 + 8g + 7
__device__ __forceinline__ v16bf load_frag(const __bf16* __restrict__ base,
                                           int row, int ld, int k0, int g) {
  Frag f;
  const __bf16* p = base + (size_t)row * ld + k0 + 8 * g;
  f.q[0] = *reinterpret_cast<const uint4*>(p);
  f.q[1] = *reinterpret_cast<const uint4*>(p + 16);
  return f.v;
}

// ---------------------------------------------------------------------------
// Generic wave-per-tile bf16 GEMM:  C[M,N] = A[M,K] * W[N,K]^T   (f32 out)
// M,N multiples of 16; K multiple of 32.
// ---------------------------------------------------------------------------
__global__ void k_gemm_bf16(const __bf16* __restrict__ A,
                            const __bf16* __restrict__ W,
                            float* __restrict__ C,
                            int M, int N, int K, int ldc) {
  int wave = (int)((blockIdx.x * blockDim.x + threadIdx.x) >> 5);
  int lane = threadIdx.x & 31;
  int nt = N >> 4;
  int tiles = (M >> 4) * nt;
  if (wave >= tiles) return;                 // wave-uniform; EXEC stays all-1s
  int m0 = (wave / nt) << 4;
  int n0 = (wave % nt) << 4;
  int ln = lane & 15;
  int g  = lane >> 4;

  v8f acc = {};
  for (int k0 = 0; k0 < K; k0 += 32) {
    v16bf a = load_frag(A, m0 + ln, K, k0, g);
    v16bf b = load_frag(W, n0 + ln, K, k0, g);
    acc = __builtin_amdgcn_wmma_f32_16x16x32_bf16(false, a, false, b,
                                                  (short)0, acc, false, false);
  }
  float* cp = C + (size_t)(m0 + 8 * g) * ldc + n0 + ln;
#pragma unroll
  for (int r = 0; r < 8; ++r) cp[(size_t)r * ldc] = acc[r];
}

// ---------------------------------------------------------------------------
// Prep kernels
// ---------------------------------------------------------------------------
__global__ void k_f2bf(const float* __restrict__ s, __bf16* __restrict__ d, int n) {
  int i = blockIdx.x * blockDim.x + threadIdx.x;
  if (i < n) d[i] = f2bf(s[i]);
}

// Build padded concatenated x_proj weight: wcat[160][192], rows >= 152 are zero.
__global__ void k_wcat(const float* __restrict__ xpw, __bf16* __restrict__ wcat) {
  int i = blockIdx.x * blockDim.x + threadIdx.x;   // 160*192
  if (i >= XDLD * DI) return;
  int row = i / DI, c = i - row * DI;
  float v = (row < KD * 38) ? xpw[(size_t)row * DI + c] : 0.0f;
  wcat[i] = f2bf(v);
}

__global__ void k_aneg(const float* __restrict__ alogs, float* __restrict__ an, int n) {
  int i = blockIdx.x * blockDim.x + threadIdx.x;
  if (i < n) an[i] = -__expf(alogs[i]);
}

// ---------------------------------------------------------------------------
// Depthwise 3x3 SAME conv + bias + SiLU over NHWC. Input = xz[:, :, 0:192].
// Emits f32 (for scan u) and bf16 (A operand of x_proj GEMM).
// ---------------------------------------------------------------------------
__global__ void k_conv(const float* __restrict__ xz,
                       const float* __restrict__ cw,
                       const float* __restrict__ cb,
                       float* __restrict__ xc, __bf16* __restrict__ xcb) {
  int idx = blockIdx.x * blockDim.x + threadIdx.x;      // B*L*DI
  if (idx >= BB * LL * DI) return;
  int c    = idx % DI;
  int rest = idx / DI;
  int pos  = rest & (LL - 1);
  int b    = rest >> 12;
  int h = pos >> 6, w = pos & 63;
  float acc = cb[c];
#pragma unroll
  for (int kh = 0; kh < 3; ++kh) {
    int hh = h + kh - 1;
    if (hh < 0 || hh > 63) continue;
#pragma unroll
    for (int kw = 0; kw < 3; ++kw) {
      int ww = w + kw - 1;
      if (ww < 0 || ww > 63) continue;
      acc = fmaf(xz[((size_t)(b * LL + (hh << 6) + ww)) * 384 + c],
                 cw[c * 9 + kh * 3 + kw], acc);
    }
  }
  float s = acc / (1.0f + __expf(-acc));                 // SiLU
  xc[idx]  = s;
  xcb[idx] = f2bf(s);
}

// ---------------------------------------------------------------------------
// Selective scan, phase 1: per-chunk local state + decay products.
// grid = (CH, K, B), block = 192 (one thread per channel).
// ---------------------------------------------------------------------------
__global__ void k_scan1(const float* __restrict__ xdbl,
                        const float* __restrict__ xc,
                        const int*   __restrict__ scan_ids,
                        const float* __restrict__ aneg,
                        const float* __restrict__ dtw,
                        const float* __restrict__ dtb,
                        float* __restrict__ hloc, float* __restrict__ pbuf) {
  int chunk = blockIdx.x, k = blockIdx.y, b = blockIdx.z;
  int c = threadIdx.x;
  __shared__ int   spos[CT];
  __shared__ float sh[CT * 38];
  if (c < CT) spos[c] = scan_ids[k * LL + chunk * CT + c];
  __syncthreads();
  for (int i = c; i < CT * 38; i += DI) {
    int t = i / 38, d = i - t * 38;
    sh[i] = xdbl[(size_t)(b * LL + spos[t]) * XDLD + k * 38 + d];
  }
  __syncthreads();

  float Ar[NS], hA[NS], Pv[NS], dw[RK];
#pragma unroll
  for (int n = 0; n < NS; ++n) {
    Ar[n] = aneg[(size_t)(k * DI + c) * NS + n];
    hA[n] = 0.0f;
    Pv[n] = 1.0f;
  }
#pragma unroll
  for (int r = 0; r < RK; ++r) dw[r] = dtw[(size_t)(k * DI + c) * RK + r];
  float db = dtb[k * DI + c];

  for (int t = 0; t < CT; ++t) {
    const float* row = &sh[t * 38];
    float dr = db;
#pragma unroll
    for (int r = 0; r < RK; ++r) dr = fmaf(dw[r], row[r], dr);
    float delta = (dr > 20.0f) ? dr : log1pf(__expf(dr));
    float u  = xc[(size_t)(b * LL + spos[t]) * DI + c];
    float du = delta * u;
#pragma unroll
    for (int n = 0; n < NS; ++n) {
      float a = __expf(delta * Ar[n]);
      hA[n] = fmaf(a, hA[n], du * row[RK + n]);
      Pv[n] *= a;
    }
  }
  size_t base = ((size_t)((b * KD + k) * CH + chunk)) * (DI * NS) + (size_t)c * NS;
#pragma unroll
  for (int n = 0; n < NS; ++n) { hloc[base + n] = hA[n]; pbuf[base + n] = Pv[n]; }
}

// Phase 2: sequential combine across chunks -> per-chunk start states.
__global__ void k_scan2(const float* __restrict__ hloc,
                        const float* __restrict__ pbuf,
                        float* __restrict__ hstart) {
  int idx = blockIdx.x * blockDim.x + threadIdx.x;   // B*K*DI*NS = 49152
  if (idx >= BB * KD * DI * NS) return;
  int cn = idx % (DI * NS);
  int bk = idx / (DI * NS);
  float carry = 0.0f;
  for (int j = 0; j < CH; ++j) {
    size_t off = ((size_t)(bk * CH + j)) * (DI * NS) + cn;
    hstart[off] = carry;
    carry = fmaf(pbuf[off], carry, hloc[off]);
  }
}

// Phase 3: re-scan each chunk from its true start state; emit y scattered to
// original spatial positions (inverse gather is free) + spatial-sum atomics.
__global__ void k_scan3(const float* __restrict__ xdbl,
                        const float* __restrict__ xc,
                        const int*   __restrict__ scan_ids,
                        const float* __restrict__ aneg,
                        const float* __restrict__ dtw,
                        const float* __restrict__ dtb,
                        const float* __restrict__ Ds,
                        const float* __restrict__ hstart,
                        float* __restrict__ yall, float* __restrict__ sbuf) {
  int chunk = blockIdx.x, k = blockIdx.y, b = blockIdx.z;
  int c = threadIdx.x;
  __shared__ int   spos[CT];
  __shared__ float sh[CT * 38];
  if (c < CT) spos[c] = scan_ids[k * LL + chunk * CT + c];
  __syncthreads();
  for (int i = c; i < CT * 38; i += DI) {
    int t = i / 38, d = i - t * 38;
    sh[i] = xdbl[(size_t)(b * LL + spos[t]) * XDLD + k * 38 + d];
  }
  __syncthreads();

  size_t hbase = ((size_t)((b * KD + k) * CH + chunk)) * (DI * NS) + (size_t)c * NS;
  float Ar[NS], hA[NS], dw[RK];
#pragma unroll
  for (int n = 0; n < NS; ++n) {
    Ar[n] = aneg[(size_t)(k * DI + c) * NS + n];
    hA[n] = hstart[hbase + n];
  }
#pragma unroll
  for (int r = 0; r < RK; ++r) dw[r] = dtw[(size_t)(k * DI + c) * RK + r];
  float db = dtb[k * DI + c];
  float Dv = Ds[k * DI + c];
  float sacc = 0.0f;

  for (int t = 0; t < CT; ++t) {
    const float* row = &sh[t * 38];
    float dr = db;
#pragma unroll
    for (int r = 0; r < RK; ++r) dr = fmaf(dw[r], row[r], dr);
    float delta = (dr > 20.0f) ? dr : log1pf(__expf(dr));
    float u  = xc[(size_t)(b * LL + spos[t]) * DI + c];
    float du = delta * u;
    float y  = u * Dv;
#pragma unroll
    for (int n = 0; n < NS; ++n) {
      float a = __expf(delta * Ar[n]);
      hA[n] = fmaf(a, hA[n], du * row[RK + n]);
      y = fmaf(hA[n], row[RK + NS + n], y);
    }
    yall[((size_t)(b * KD + k) * LL + spos[t]) * DI + c] = y;
    sacc += y;
  }
  atomicAdd(&sbuf[(b * KD + k) * DI + c], sacc);
}

// Gate: gscale[b, k*DI+c] = sigmoid( sum_i mean_s[b,i*DI+c] * Wg[c][k][i] )
__global__ void k_gate(const float* __restrict__ sbuf,
                       const float* __restrict__ gw,
                       float* __restrict__ gscale) {
  int b = blockIdx.x;
  int t = threadIdx.x;                 // 768
  int k = t / DI, c = t - k * DI;
  float acc = 0.0f;
#pragma unroll
  for (int i = 0; i < KD; ++i)
    acc = fmaf(sbuf[b * (KD * DI) + i * DI + c] * (1.0f / (float)LL),
               gw[(c * 4 + k) * 4 + i], acc);
  gscale[b * (KD * DI) + t] = 1.0f / (1.0f + __expf(-acc));
}

// Fuse: gated direction-sum + LayerNorm(192) + SiLU(z) gating -> bf16 for out_proj.
__global__ void k_fuse(const float* __restrict__ yall,
                       const float* __restrict__ gscale,
                       const float* __restrict__ xz,
                       const float* __restrict__ lng,
                       const float* __restrict__ lnb,
                       __bf16* __restrict__ ynb) {
  int row = blockIdx.x;                // B*L
  int c   = threadIdx.x;               // 192
  int b   = row >> 12;
  int pos = row & (LL - 1);
  float ys = 0.0f;
#pragma unroll
  for (int k = 0; k < KD; ++k)
    ys = fmaf(yall[((size_t)(b * KD + k) * LL + pos) * DI + c],
              gscale[b * (KD * DI) + k * DI + c], ys);

  __shared__ float s1[DI], s2[DI];
  s1[c] = ys; s2[c] = ys * ys;
  __syncthreads();
  if (c < 64) {
    s1[c] += s1[c + 64] + s1[c + 128];
    s2[c] += s2[c + 64] + s2[c + 128];
  }
  __syncthreads();
  for (int st = 32; st > 0; st >>= 1) {
    if (c < st) { s1[c] += s1[c + st]; s2[c] += s2[c + st]; }
    __syncthreads();
  }
  float mean = s1[0] * (1.0f / DI);
  float var  = s2[0] * (1.0f / DI) - mean * mean;
  float yn = (ys - mean) * rsqrtf(var + 1e-5f) * lng[c] + lnb[c];
  float z  = xz[(size_t)row * 384 + DI + c];
  yn *= z / (1.0f + __expf(-z));       // * silu(z)
  ynb[(size_t)row * DI + c] = f2bf(yn);
}

// ---------------------------------------------------------------------------
// Host launcher
// ---------------------------------------------------------------------------
static inline size_t align256(size_t x) { return (x + 255) & ~(size_t)255; }

extern "C" void kernel_launch(void* const* d_in, const int* in_sizes, int n_in,
                              void* d_out, int out_size, void* d_ws, size_t ws_size,
                              hipStream_t stream) {
  (void)in_sizes; (void)n_in; (void)out_size; (void)ws_size;
  const float* x        = (const float*)d_in[0];   // (4,64,64,96)
  const float* in_w     = (const float*)d_in[1];   // (384,96)
  const float* conv_w   = (const float*)d_in[2];   // (192,1,3,3)
  const float* conv_b   = (const float*)d_in[3];   // (192)
  const float* xproj_w  = (const float*)d_in[4];   // (4,38,192)
  const float* dt_w     = (const float*)d_in[5];   // (4,192,6)
  const float* dt_b     = (const float*)d_in[6];   // (4,192)
  const float* A_logs   = (const float*)d_in[7];   // (768,16)
  const float* Ds       = (const float*)d_in[8];   // (768)
  const float* ln_g     = (const float*)d_in[9];   // (192)
  const float* ln_b     = (const float*)d_in[10];  // (192)
  const float* out_w    = (const float*)d_in[11];  // (96,192)
  const float* gate_w   = (const float*)d_in[12];  // (768,4)
  const int*   scan_ids = (const int*)d_in[13];    // (4,4096)
  float* out = (float*)d_out;                      // (4,64,64,96)

  // Workspace layout
  char* p = (char*)d_ws;
  size_t off = 0;
  auto take = [&](size_t bytes) { void* r = p + off; off += align256(bytes); return r; };

  __bf16* xbf    = (__bf16*)take((size_t)ROWS * DM * 2);       // x in bf16
  __bf16* inwbf  = (__bf16*)take((size_t)384 * DM * 2);
  __bf16* wcat   = (__bf16*)take((size_t)XDLD * DI * 2);
  __bf16* outwbf = (__bf16*)take((size_t)DM * DI * 2);
  float*  aneg   = (float*) take((size_t)KD * DI * NS * 4);
  float*  xz     = (float*) take((size_t)ROWS * 384 * 4);
  float*  xc     = (float*) take((size_t)ROWS * DI * 4);
  __bf16* xcb    = (__bf16*)take((size_t)ROWS * DI * 2);
  float*  xdbl   = (float*) take((size_t)ROWS * XDLD * 4);
  float*  hloc   = (float*) take((size_t)BB * KD * CH * DI * NS * 4);
  float*  pbuf   = (float*) take((size_t)BB * KD * CH * DI * NS * 4);
  float*  hstart = (float*) take((size_t)BB * KD * CH * DI * NS * 4);
  float*  yall   = (float*) take((size_t)BB * KD * LL * DI * 4);
  float*  sbuf   = (float*) take((size_t)BB * KD * DI * 4);
  float*  gscale = (float*) take((size_t)BB * KD * DI * 4);
  __bf16* ynb    = (__bf16*)take((size_t)ROWS * DI * 2);

  // ---- Prep / conversions ----
  {
    int n = ROWS * DM;
    k_f2bf<<<(n + 255) / 256, 256, 0, stream>>>(x, xbf, n);
  }
  {
    int n = 384 * DM;
    k_f2bf<<<(n + 255) / 256, 256, 0, stream>>>(in_w, inwbf, n);
  }
  {
    int n = XDLD * DI;
    k_wcat<<<(n + 255) / 256, 256, 0, stream>>>(xproj_w, wcat);
  }
  {
    int n = DM * DI;
    k_f2bf<<<(n + 255) / 256, 256, 0, stream>>>(out_w, outwbf, n);
  }
  {
    int n = KD * DI * NS;
    k_aneg<<<(n + 255) / 256, 256, 0, stream>>>(A_logs, aneg, n);
  }

  // ---- GEMM 1: xz = x @ in_proj_w^T  (16384 x 384, K=96) ----
  {
    int tiles = (ROWS / 16) * (384 / 16);
    k_gemm_bf16<<<(tiles + 7) / 8, 256, 0, stream>>>(xbf, inwbf, xz,
                                                     ROWS, 384, DM, 384);
  }

  // ---- Depthwise conv + SiLU ----
  {
    int n = BB * LL * DI;
    k_conv<<<(n + 255) / 256, 256, 0, stream>>>(xz, conv_w, conv_b, xc, xcb);
  }

  // ---- GEMM 2: x_dbl = xc @ wcat^T  (16384 x 160, K=192) ----
  {
    int tiles = (ROWS / 16) * (XDLD / 16);
    k_gemm_bf16<<<(tiles + 7) / 8, 256, 0, stream>>>(xcb, wcat, xdbl,
                                                     ROWS, XDLD, DI, XDLD);
  }

  // ---- Chunked selective scan ----
  hipMemsetAsync(sbuf, 0, (size_t)BB * KD * DI * sizeof(float), stream);
  {
    dim3 g(CH, KD, BB);
    k_scan1<<<g, DI, 0, stream>>>(xdbl, xc, scan_ids, aneg, dt_w, dt_b, hloc, pbuf);
  }
  {
    int n = BB * KD * DI * NS;
    k_scan2<<<(n + 255) / 256, 256, 0, stream>>>(hloc, pbuf, hstart);
  }
  {
    dim3 g(CH, KD, BB);
    k_scan3<<<g, DI, 0, stream>>>(xdbl, xc, scan_ids, aneg, dt_w, dt_b, Ds,
                                  hstart, yall, sbuf);
  }

  // ---- Gate + fuse (LN, SiLU(z)) ----
  k_gate<<<BB, KD * DI, 0, stream>>>(sbuf, gate_w, gscale);
  k_fuse<<<ROWS, DI, 0, stream>>>(yall, gscale, xz, ln_g, ln_b, ynb);

  // ---- GEMM 3: out = yn @ out_proj_w^T  (16384 x 96, K=192) ----
  {
    int tiles = (ROWS / 16) * (DM / 16);
    k_gemm_bf16<<<(tiles + 7) / 8, 256, 0, stream>>>(ynb, outwbf, out,
                                                     ROWS, DM, DI, DM);
  }
}
